// LlamaAttention_55465207660620
// MI455X (gfx1250) — matrix operand order repaired and checked by reading, not compiled, not run on hardware
//
#include <hip/hip_runtime.h>

// ---------------- problem constants ----------------
#define B_SZ  2
#define L_SEQ 2048
#define DM    2048
#define NH    32
#define NKV   8
#define HD    64
#define DKV   (NKV * HD)        // 512
#define MTOK  (B_SZ * L_SEQ)    // 4096

typedef __bf16 bf16_t;
typedef __bf16 v16bf __attribute__((ext_vector_type(16)));
typedef __bf16 v8bf  __attribute__((ext_vector_type(8)));
typedef float  v8f   __attribute__((ext_vector_type(8)));
typedef int    v4i   __attribute__((ext_vector_type(4)));

// ---------------- CDNA5 async global->LDS path (guarded) ----------------
#if defined(__has_builtin)
#if __has_builtin(__builtin_amdgcn_global_load_async_to_lds_b128) && \
    __has_builtin(__builtin_amdgcn_s_wait_asynccnt)
#define HAVE_ASYNC_LDS 1
#endif
#endif

__device__ __forceinline__ void copy16_async(const bf16_t* g, bf16_t* l) {
#ifdef HAVE_ASYNC_LDS
    __builtin_amdgcn_global_load_async_to_lds_b128(
        (__attribute__((address_space(1))) v4i*)const_cast<bf16_t*>(g),
        (__attribute__((address_space(3))) v4i*)l,
        /*imm offset=*/0, /*cpol=*/0);
#else
    *(v8bf*)l = *(const v8bf*)g;
#endif
}

#ifdef HAVE_ASYNC_LDS
#define ASYNC_WAIT(n) __builtin_amdgcn_s_wait_asynccnt(n)
#else
#define ASYNC_WAIT(n) ((void)0)
#endif

// ---------------- helpers ----------------
__device__ __forceinline__ bf16_t f2bf(float f) {
    unsigned u = __float_as_uint(f);
    unsigned r = u + 0x7FFFu + ((u >> 16) & 1u);   // round-to-nearest-even
    unsigned short h = (unsigned short)(r >> 16);
    bf16_t b;
    __builtin_memcpy(&b, &h, 2);
    return b;
}

__device__ __forceinline__ v16bf make_frag(v8bf lo, v8bf hi) {
    v16bf f;
#pragma unroll
    for (int i = 0; i < 8; ++i) { f[i] = lo[i]; f[i + 8] = hi[i]; }
    return f;
}

__device__ __forceinline__ v8f wmma_bf16(v16bf a, v16bf b, v8f c) {
    // D(16x16,f32) = A(16x32,bf16) * B(32x16,bf16) + C
    return __builtin_amdgcn_wmma_f32_16x16x32_bf16(
        /*neg_a=*/false, a, /*neg_b=*/false, b,
        /*c_mod=*/(short)0, c, /*reuse_a=*/false, /*reuse_b=*/false);
}

// ---------------- elementwise conversions ----------------
__global__ void conv_f32_bf16_kernel(const float* __restrict__ in,
                                     bf16_t* __restrict__ out, int total) {
    int idx = blockIdx.x * blockDim.x + threadIdx.x;
    if (idx < total) out[idx] = f2bf(in[idx]);
}

// Wt[k*N + n] = bf16(W[n*K + k])   (W is (N,K) row-major; Wt is (K,N) row-major)
__global__ void wtrans_kernel(const float* __restrict__ W,
                              bf16_t* __restrict__ Wt, int N, int K, int total) {
    int idx = blockIdx.x * blockDim.x + threadIdx.x;
    if (idx >= total) return;
    int k = idx % K;
    int n = idx / K;
    Wt[(size_t)k * N + n] = f2bf(W[idx]);
}

// ---------------- bf16 WMMA GEMM with async LDS double-buffering ----------------
// C(f32, MxN) = A(bf16, MxK) @ Bt(bf16, KxN). Block: 256 thr = 8 waves,
// tile 64(M) x 128(N); per k-step the block stages A(64x32)+B(32x128) in LDS.
template <int N, int K>
__global__ void __launch_bounds__(256)
gemm_bf16_kernel(const bf16_t* __restrict__ A, const bf16_t* __restrict__ Bt,
                 float* __restrict__ C) {
    __shared__ __align__(16) bf16_t As[2][64][32];    //  8 KB
    __shared__ __align__(16) bf16_t Bs[2][32][128];   // 16 KB

    const int t    = threadIdx.x;
    const int lane = t & 31;
    const int wave = t >> 5;
    const int wm = wave & 1;       // 2 waves along M
    const int wn = wave >> 1;      // 4 waves along N
    const int m0b = blockIdx.y * 64;
    const int n0b = blockIdx.x * 128;

    // staging assignments (per thread: 1x16B for A, 2x16B for B)
    const int arow = t >> 2;          // 0..63
    const int acol = (t & 3) * 8;     // 0,8,16,24
    const int brow = t >> 3;          // 0..31
    const int bcol = (t & 7) * 16;    // 0..112

    const int rowl = lane & 15;
    const int koff = (lane >> 4) * 8;

    v8f acc[2][2] = {};

    const int KT = K / 32;

    // prefetch tile 0
    copy16_async(A + (size_t)(m0b + arow) * K + acol, &As[0][arow][acol]);
    copy16_async(Bt + (size_t)brow * N + n0b + bcol,     &Bs[0][brow][bcol]);
    copy16_async(Bt + (size_t)brow * N + n0b + bcol + 8, &Bs[0][brow][bcol + 8]);

    for (int kt = 0; kt < KT; ++kt) {
        const int cur = kt & 1;
        if (kt + 1 < KT) {
            const int nxt = cur ^ 1;
            const int kn = (kt + 1) * 32;
            copy16_async(A + (size_t)(m0b + arow) * K + kn + acol,
                         &As[nxt][arow][acol]);
            copy16_async(Bt + (size_t)(kn + brow) * N + n0b + bcol,
                         &Bs[nxt][brow][bcol]);
            copy16_async(Bt + (size_t)(kn + brow) * N + n0b + bcol + 8,
                         &Bs[nxt][brow][bcol + 8]);
            ASYNC_WAIT(3);   // current tile's 3 ops done; next tile in flight
        } else {
            ASYNC_WAIT(0);
        }
        __syncthreads();     // whole block's staging of `cur` complete

        v16bf af[2], bfr[2];
#pragma unroll
        for (int i = 0; i < 2; ++i) {
            const bf16_t* ap = &As[cur][wm * 32 + i * 16 + rowl][koff];
            af[i] = make_frag(*(const v8bf*)ap, *(const v8bf*)(ap + 16));
        }
#pragma unroll
        for (int j = 0; j < 2; ++j) {
            // B layout: lane%16 <-> N, elements <-> K (table order)
            for (int s = 0; s < 16; ++s)
                bfr[j][s] = Bs[cur][(s < 8 ? koff + s : 16 + koff + s - 8)][wn * 32 + j * 16 + rowl];
        }
#pragma unroll
        for (int i = 0; i < 2; ++i)
#pragma unroll
            for (int j = 0; j < 2; ++j)
                acc[i][j] = wmma_bf16(af[i], bfr[j], acc[i][j]);

        __syncthreads();     // done reading `cur` before it is overwritten
    }

    // C layout: lane l, vgpr r -> (m = r + 8*(l>=16), n = l&15)
    const int mrow = (lane >> 4) * 8;
    const int ncol = lane & 15;
#pragma unroll
    for (int i = 0; i < 2; ++i)
#pragma unroll
        for (int j = 0; j < 2; ++j) {
            float* cp = C + (size_t)(m0b + wm * 32 + i * 16 + mrow) * N +
                        n0b + wn * 32 + j * 16 + ncol;
#pragma unroll
            for (int r = 0; r < 8; ++r)
                cp[(size_t)r * N] = acc[i][j][r];
        }
}

// ---------------- RoPE + layout kernels ----------------
// Q: qraw (B,L,2048) f32 -> Q (B,H,L,64) bf16 with RoPE
__global__ void rope_q_kernel(const float* __restrict__ qraw,
                              const float* __restrict__ cosT,
                              const float* __restrict__ sinT,
                              bf16_t* __restrict__ Qo, int total) {
    int idx = blockIdx.x * blockDim.x + threadIdx.x;
    if (idx >= total) return;
    int i = idx & 63;
    int t = idx >> 6;
    int l = t & (L_SEQ - 1);
    int t2 = t >> 11;
    int h = t2 & (NH - 1);
    int b = t2 >> 5;
    size_t base = ((size_t)(b * L_SEQ + l)) * DM + h * HD;
    float q = qraw[base + i];
    float rot = (i < 32) ? -qraw[base + i + 32] : qraw[base + i - 32];
    float val = q * cosT[l * HD + i] + rot * sinT[l * HD + i];
    Qo[(((size_t)(b * NH + h)) * L_SEQ + l) * HD + i] = f2bf(val);
}

// K: kraw (B,L,512) f32 -> Kt (B,Hk,64,L) bf16 (d-major, transposed) with RoPE
__global__ void rope_k_kernel(const float* __restrict__ kraw,
                              const float* __restrict__ cosT,
                              const float* __restrict__ sinT,
                              bf16_t* __restrict__ Kt, int total) {
    int idx = blockIdx.x * blockDim.x + threadIdx.x;
    if (idx >= total) return;
    int i = idx & 63;
    int t = idx >> 6;
    int l = t & (L_SEQ - 1);
    int t2 = t >> 11;
    int hk = t2 & (NKV - 1);
    int b = t2 >> 3;
    size_t base = ((size_t)(b * L_SEQ + l)) * DKV + hk * HD;
    float k = kraw[base + i];
    float rot = (i < 32) ? -kraw[base + i + 32] : kraw[base + i - 32];
    float val = k * cosT[l * HD + i] + rot * sinT[l * HD + i];
    Kt[(((size_t)(b * NKV + hk)) * HD + i) * L_SEQ + l] = f2bf(val);
}

// V: vraw (B,L,512) f32 -> V (B,Hk,L,64) bf16
__global__ void conv_v_kernel(const float* __restrict__ vraw,
                              bf16_t* __restrict__ Vo, int total) {
    int idx = blockIdx.x * blockDim.x + threadIdx.x;
    if (idx >= total) return;
    int i = idx & 63;
    int t = idx >> 6;
    int l = t & (L_SEQ - 1);
    int t2 = t >> 11;
    int hk = t2 & (NKV - 1);
    int b = t2 >> 3;
    float v = vraw[((size_t)(b * L_SEQ + l)) * DKV + hk * HD + i];
    Vo[(((size_t)(b * NKV + hk)) * L_SEQ + l) * HD + i] = f2bf(v);
}

// ---------------- flash attention (causal, GQA) ----------------
// grid: B*H*(L/64) blocks, 128 threads (4 waves). Each wave: 16 query rows.
__global__ void __launch_bounds__(128)
attn_kernel(const bf16_t* __restrict__ Q,   // (B,H,L,64)
            const bf16_t* __restrict__ Kt,  // (B,Hk,64,L)
            const bf16_t* __restrict__ V,   // (B,Hk,L,64)
            bf16_t* __restrict__ Ob) {      // (B,L,2048)
    __shared__ __align__(16) bf16_t Pl[4][16][32];  // wave-private P tiles

    const int lane = threadIdx.x & 31;
    const int wave = threadIdx.x >> 5;
    const int qt = blockIdx.x & 31;           // L/64 = 32 query tiles
    const int h  = (blockIdx.x >> 5) & 31;
    const int b  = blockIdx.x >> 10;
    const int kv = h >> 2;                    // GQA: 4 Q heads per KV head
    const int qrow0 = qt * 64 + wave * 16;

    const bf16_t* Qb  = Q  + ((size_t)(b * NH + h)) * L_SEQ * HD;
    const bf16_t* Ktb = Kt + ((size_t)(b * NKV + kv)) * HD * L_SEQ;
    const bf16_t* Vb  = V  + ((size_t)(b * NKV + kv)) * L_SEQ * HD;

    const int rowl = lane & 15;
    const int half = lane >> 4;
    const int koff = half * 8;

    // Q fragments for the whole tile (d = 0..31 and 32..63), reused every step
    v16bf qf[2];
#pragma unroll
    for (int dh = 0; dh < 2; ++dh) {
        const bf16_t* qp = Qb + (size_t)(qrow0 + rowl) * HD + dh * 32 + koff;
        qf[dh] = make_frag(*(const v8bf*)qp, *(const v8bf*)(qp + 16));
    }

    v8f o[4] = {};
    float rmax[8], rsum[8];
#pragma unroll
    for (int r = 0; r < 8; ++r) { rmax[r] = -3.0e38f; rsum[r] = 0.0f; }

    const float scale = 0.125f;               // 1/sqrt(64)
    const int m_hi = qrow0 + 15;

    for (int kb = 0; kb <= m_hi; kb += 32) {
        // K fragments: key chunk c (16 keys), d-half dh
        v16bf kf[2][2];
#pragma unroll
        for (int c = 0; c < 2; ++c)
#pragma unroll
            for (int dh = 0; dh < 2; ++dh) {
                for (int s = 0; s < 16; ++s)
                    kf[c][dh][s] = Ktb[(size_t)(dh * 32 + (s < 8 ? koff + s : 16 + koff + s - 8)) * L_SEQ + kb + c * 16 + rowl];
            }

        v8f s0 = {}, s1 = {};
        s0 = wmma_bf16(qf[0], kf[0][0], s0);
        s0 = wmma_bf16(qf[1], kf[0][1], s0);
        s1 = wmma_bf16(qf[0], kf[1][0], s1);
        s1 = wmma_bf16(qf[1], kf[1][1], s1);

        const int mbase = qrow0 + half * 8;
        const int c0 = kb + (lane & 15);
        const int c1 = c0 + 16;
#pragma unroll
        for (int r = 0; r < 8; ++r) {
            const int m = mbase + r;
            float a  = s0[r] * scale; if (c0 > m) a  = -1.0e30f;
            float bb = s1[r] * scale; if (c1 > m) bb = -1.0e30f;
            // row max across the 16 lanes of this half-wave
            float mx = fmaxf(a, bb);
#pragma unroll
            for (int off = 1; off < 16; off <<= 1)
                mx = fmaxf(mx, __shfl_xor(mx, off, 32));
            const float nm = fmaxf(rmax[r], mx);
            const float corr = __expf(rmax[r] - nm);
            const float p0 = __expf(a - nm);
            const float p1 = __expf(bb - nm);
            float rs = p0 + p1;
#pragma unroll
            for (int off = 1; off < 16; off <<= 1)
                rs += __shfl_xor(rs, off, 32);
            rsum[r] = rsum[r] * corr + rs;
            rmax[r] = nm;
            o[0][r] *= corr; o[1][r] *= corr; o[2][r] *= corr; o[3][r] *= corr;
            // stash P in C-layout into wave-private LDS
            Pl[wave][half * 8 + r][lane & 15]        = f2bf(p0);
            Pl[wave][half * 8 + r][16 + (lane & 15)] = f2bf(p1);
        }

        asm volatile("s_wait_dscnt 0" ::: "memory");

        // reload P in A-fragment layout (16x32)
        const bf16_t* pp = &Pl[wave][rowl][koff];
        v16bf pf = make_frag(*(const v8bf*)pp, *(const v8bf*)(pp + 16));

        // O += P @ V  (V natural layout is the (K,N) B operand)
#pragma unroll
        for (int t = 0; t < 4; ++t) {
            v16bf vf;
            for (int s = 0; s < 16; ++s)
                vf[s] = Vb[(size_t)(kb + (s < 8 ? koff + s : 16 + koff + s - 8)) * HD + t * 16 + rowl];
            o[t] = wmma_bf16(pf, vf, o[t]);
        }
    }

    // normalize + store (B,L,H*64) bf16
#pragma unroll
    for (int r = 0; r < 8; ++r) {
        const float inv = 1.0f / rsum[r];
        const int m = qrow0 + half * 8 + r;
        const size_t rowoff = ((size_t)b * L_SEQ + m) * DM + h * HD;
#pragma unroll
        for (int t = 0; t < 4; ++t)
            Ob[rowoff + t * 16 + (lane & 15)] = f2bf(o[t][r] * inv);
    }
}

// ---------------- host-side launch ----------------
extern "C" void kernel_launch(void* const* d_in, const int* in_sizes, int n_in,
                              void* d_out, int out_size, void* d_ws, size_t ws_size,
                              hipStream_t stream) {
    const float* x    = (const float*)d_in[0];
    const float* cosT = (const float*)d_in[1];
    const float* sinT = (const float*)d_in[2];
    const float* Wq   = (const float*)d_in[3];
    const float* Wk   = (const float*)d_in[4];
    const float* Wv   = (const float*)d_in[5];
    const float* Wo   = (const float*)d_in[6];
    float* out = (float*)d_out;

    char* ws = (char*)d_ws;
    // workspace layout (bytes)
    bf16_t* xb   = (bf16_t*)(ws + 0);                    // 16 MB  (4096x2048 bf16)
    bf16_t* wqt  = (bf16_t*)(ws + 16777216);             //  8 MB  (2048x2048)
    bf16_t* wkt  = (bf16_t*)(ws + 25165824);             //  2 MB  (2048x512)
    bf16_t* wvt  = (bf16_t*)(ws + 27262976);             //  2 MB
    bf16_t* wot  = (bf16_t*)(ws + 29360128);             //  8 MB
    float*  qraw = (float*)(ws + 37748736);              // 32 MB  (4096x2048 f32)
    float*  kraw = (float*)(ws + 71303168);              //  8 MB  (4096x512)
    float*  vraw = (float*)(ws + 79691776);              //  8 MB
    bf16_t* qb   = (bf16_t*)(ws + 88080384);             // 16 MB  (B,H,L,64)
    bf16_t* ktb  = (bf16_t*)(ws + 104857600);            //  4 MB  (B,Hk,64,L)
    bf16_t* vb   = (bf16_t*)(ws + 109051904);            //  4 MB  (B,Hk,L,64)
    bf16_t* ob   = (bf16_t*)(ws + 113246208);            // 16 MB  (B,L,2048)

    const int TPB = 256;

    // 1) x -> bf16
    {
        int total = MTOK * DM;
        conv_f32_bf16_kernel<<<(total + TPB - 1) / TPB, TPB, 0, stream>>>(x, xb, total);
    }
    // 2) transpose+convert weights to (K,N) bf16
    {
        int total = DM * DM;
        wtrans_kernel<<<(total + TPB - 1) / TPB, TPB, 0, stream>>>(Wq, wqt, DM, DM, total);
        wtrans_kernel<<<(total + TPB - 1) / TPB, TPB, 0, stream>>>(Wo, wot, DM, DM, total);
        int totkv = DKV * DM;
        wtrans_kernel<<<(totkv + TPB - 1) / TPB, TPB, 0, stream>>>(Wk, wkt, DKV, DM, totkv);
        wtrans_kernel<<<(totkv + TPB - 1) / TPB, TPB, 0, stream>>>(Wv, wvt, DKV, DM, totkv);
    }
    // 3) projections (WMMA GEMMs, async-LDS double buffered)
    {
        dim3 gq(DM / 128, MTOK / 64);
        gemm_bf16_kernel<DM, DM><<<gq, 256, 0, stream>>>(xb, wqt, qraw);
        dim3 gkv(DKV / 128, MTOK / 64);
        gemm_bf16_kernel<DKV, DM><<<gkv, 256, 0, stream>>>(xb, wkt, kraw);
        gemm_bf16_kernel<DKV, DM><<<gkv, 256, 0, stream>>>(xb, wvt, vraw);
    }
    // 4) RoPE + layout changes
    {
        int totq = B_SZ * NH * L_SEQ * HD;
        rope_q_kernel<<<(totq + TPB - 1) / TPB, TPB, 0, stream>>>(qraw, cosT, sinT, qb, totq);
        int totk = B_SZ * NKV * L_SEQ * HD;
        rope_k_kernel<<<(totk + TPB - 1) / TPB, TPB, 0, stream>>>(kraw, cosT, sinT, ktb, totk);
        conv_v_kernel<<<(totk + TPB - 1) / TPB, TPB, 0, stream>>>(vraw, vb, totk);
    }
    // 5) flash attention (WMMA)
    {
        int blocks = B_SZ * NH * (L_SEQ / 64);   // 2048
        attn_kernel<<<blocks, 128, 0, stream>>>(qb, ktb, vb, ob);
    }
    // 6) output projection (WMMA GEMM) straight to d_out
    {
        dim3 go(DM / 128, MTOK / 64);
        gemm_bf16_kernel<DM, DM><<<go, 256, 0, stream>>>(ob, wot, out);
    }
    (void)in_sizes; (void)n_in; (void)out_size; (void)ws_size;
}
